// NTXent_84035330114174
// MI455X (gfx1250) — matrix operand order repaired and checked by reading, not compiled, not run on hardware
//
#include <hip/hip_runtime.h>
#include <cstdint>

// ---------------------------------------------------------------------------
// NT-Xent fused loss for MI455X (gfx1250, wave32, WMMA bf16)
//   N = 4096, D = 256, M = 2N = 8192, T = 0.1
// ---------------------------------------------------------------------------

typedef __attribute__((ext_vector_type(16))) __bf16 v16bf;
typedef __attribute__((ext_vector_type(8)))  float  v8f;

#define NROWS   8192          // M = 2N
#define HALF_M  4096          // N
#define DWPR    128           // dwords per bf16 row (256 bf16 = 128 u32)
#define CSLICE  4             // column slices (grid.y)
#define NTILES  128           // 2048 cols per slice / 16

// combined scale: (1/T) * log2(e) = 10 * 1.4426950408889634
#define EXP2_SCALE 14.426950408889634f

union ABf {
    v16bf    v;
    uint32_t u[8];
    uint4    q[2];
};

__device__ __forceinline__ uint32_t bfpack2(float lo, float hi) {
    uint32_t ul = __float_as_uint(lo), uh = __float_as_uint(hi);
    ul = (ul + 0x7FFFu + ((ul >> 16) & 1u)) >> 16;   // RNE f32 -> bf16
    uh = (uh + 0x7FFFu + ((uh >> 16) & 1u)) >> 16;
    return ul | (uh << 16);
}

// ---------------------------------------------------------------------------
// Kernel 1: per-row L2 normalize; emit bf16 matrix (row-major) + 1/norm table.
// One wave32 per row; lane covers 8 contiguous floats.
// ---------------------------------------------------------------------------
__global__ __launch_bounds__(256)
void ntx_normalize(const float* __restrict__ z1, const float* __restrict__ z2,
                   uint32_t* __restrict__ znbf, float* __restrict__ invn) {
    const int tid  = threadIdx.x;
    const int lane = tid & 31;
    const int w    = tid >> 5;
    const int row  = blockIdx.x * 8 + w;

    const float* src = (row < HALF_M) ? (z1 + (size_t)row * 256)
                                      : (z2 + (size_t)(row - HALF_M) * 256);
    float4 a = *(const float4*)(src + lane * 8);
    float4 b = *(const float4*)(src + lane * 8 + 4);

    float ss = a.x*a.x + a.y*a.y + a.z*a.z + a.w*a.w
             + b.x*b.x + b.y*b.y + b.z*b.z + b.w*b.w;
    #pragma unroll
    for (int m = 16; m >= 1; m >>= 1) ss += __shfl_xor(ss, m, 32);

    const float inv = 1.0f / fmaxf(sqrtf(ss), 1e-8f);
    if (lane == 0) invn[row] = inv;

    uint4 o;
    o.x = bfpack2(a.x * inv, a.y * inv);
    o.y = bfpack2(a.z * inv, a.w * inv);
    o.z = bfpack2(b.x * inv, b.y * inv);
    o.w = bfpack2(b.z * inv, b.w * inv);
    *(uint4*)(znbf + (size_t)row * DWPR + lane * 4) = o;
}

// ---------------------------------------------------------------------------
// Kernel 2: fused GEMM (bf16 WMMA) + streaming exp-sum.
//   grid = (64, CSLICE), block = 256 (8 waves).
//   Wave w owns 16 rows: A tile (16x256 bf16) held in 64 VGPRs.
//   Workgroup walks 2048 columns (128 tiles of 16); B tile staged through a
//   double-buffered 2x8KB LDS window so global->LDS latency hides under the
//   WMMA + exp work of the previous tile. Diagonal (self-similarity) masked
//   to contribute exp = 0. exp(sim/T) computed with the raw v_exp_f32
//   (hardware flush below 2^-126 is exact enough for a sum of e^[-10,10]).
// ---------------------------------------------------------------------------
__global__ __launch_bounds__(256)
void ntx_gemm_expsum(const uint32_t* __restrict__ znbf,
                     float* __restrict__ partial) {
    __shared__ __align__(32) uint32_t ldsB[2][16 * DWPR];   // 2 x 8 KB

    const int tid  = threadIdx.x;
    const int lane = tid & 31;
    const int w    = tid >> 5;
    const int half = lane >> 4;           // lane half selects K sub-range
    const int ln16 = lane & 15;
    const int rowbase = (blockIdx.x * 8 + w) * 16;
    const int cs      = blockIdx.y;
    const int colorg  = cs * 2048;

    // --- load A tile (16 rows x 256 K) into registers, ISA §7.12.2 layout ---
    const uint32_t* arow = znbf + (size_t)(rowbase + ln16) * DWPR;
    ABf a[8];
    #pragma unroll
    for (int k = 0; k < 8; ++k) {
        a[k].q[0] = *(const uint4*)(arow + 16 * k + 4 * half);
        a[k].q[1] = *(const uint4*)(arow + 16 * k + 4 * half + 8);
    }

    float sums[8] = {0.f, 0.f, 0.f, 0.f, 0.f, 0.f, 0.f, 0.f};

    // prime buffer 0 with tile 0
    #pragma unroll
    for (int i = 0; i < 8; ++i) {
        const int idx = tid + (i << 8);
        ldsB[0][idx] = znbf[(size_t)(colorg + (idx >> 7)) * DWPR + (idx & 127)];
    }

    for (int t = 0; t < NTILES; ++t) {
        __syncthreads();   // cur buffer filled; prev reads of next buffer done

        // prefetch next tile into the other buffer (uniform branch)
        if (t + 1 < NTILES) {
            const int nb = (t + 1) & 1;
            const int cb = colorg + (t + 1) * 16;
            #pragma unroll
            for (int i = 0; i < 8; ++i) {
                const int idx = tid + (i << 8);
                ldsB[nb][idx] =
                    znbf[(size_t)(cb + (idx >> 7)) * DWPR + (idx & 127)];
            }
        }

        // B chunks from the current buffer (ds_load_b128 pairs feed the WMMAs)
        const uint32_t* lb = &ldsB[t & 1][ln16 * DWPR + 4 * half];
        ABf b[8];
        #pragma unroll
        for (int k = 0; k < 8; ++k) {
            b[k].q[0] = *(const uint4*)(lb + 16 * k);
            b[k].q[1] = *(const uint4*)(lb + 16 * k + 8);
        }

        v8f acc = {};
        #pragma unroll
        for (int k = 0; k < 8; ++k)
            acc = __builtin_amdgcn_wmma_f32_16x16x32_bf16(
                false, a[k].v, false, b[k].v, (short)0, acc, false, false);

        const bool diag = ((colorg + t * 16) == rowbase);   // uniform per wave
        #pragma unroll
        for (int r = 0; r < 8; ++r) {
            // raw v_exp_f32: exp(sim/T) = 2^(sim * 10 * log2 e)
            float e = __builtin_amdgcn_exp2f(acc[r] * EXP2_SCALE);
            if (diag && (ln16 == r + 8 * half)) e = 0.f;    // mask self-sim
            sums[r] += e;
        }
    }

    // reduce each row-slot across the 16 lanes of its half-wave
    #pragma unroll
    for (int r = 0; r < 8; ++r) {
        #pragma unroll
        for (int m = 1; m <= 8; m <<= 1) sums[r] += __shfl_xor(sums[r], m, 32);
    }
    if (ln16 == 0) {
        const int rb = rowbase + 8 * half;  // lane 0 -> rows 0..7, lane 16 -> 8..15
        #pragma unroll
        for (int r = 0; r < 8; ++r)
            partial[(size_t)(rb + r) * CSLICE + cs] = sums[r];
    }
}

// ---------------------------------------------------------------------------
// Kernel 3: finalize. Single block of 256 threads (8 waves).
//   pos pair recomputed in f32 from raw z (full precision);
//   loss_i = log(sum_exp_i) - pos_i ; out = mean.
// ---------------------------------------------------------------------------
__global__ __launch_bounds__(256)
void ntx_finalize(const float* __restrict__ z1, const float* __restrict__ z2,
                  const float* __restrict__ invn,
                  const float* __restrict__ partial,
                  float* __restrict__ out) {
    __shared__ float red[8];
    const int tid  = threadIdx.x;
    const int lane = tid & 31;
    const int w    = tid >> 5;

    float acc = 0.f;
    for (int i = 0; i < NROWS / 8; ++i) {
        const int row = i * 8 + w;
        const int j   = (row + HALF_M) & (NROWS - 1);
        const float* zi = (row < HALF_M) ? (z1 + (size_t)row * 256)
                                         : (z2 + (size_t)(row - HALF_M) * 256);
        const float* zj = (j < HALF_M) ? (z1 + (size_t)j * 256)
                                       : (z2 + (size_t)(j - HALF_M) * 256);
        float4 xa = *(const float4*)(zi + lane * 8);
        float4 xb = *(const float4*)(zi + lane * 8 + 4);
        float4 ya = *(const float4*)(zj + lane * 8);
        float4 yb = *(const float4*)(zj + lane * 8 + 4);
        float p = xa.x*ya.x + xa.y*ya.y + xa.z*ya.z + xa.w*ya.w
                + xb.x*yb.x + xb.y*yb.y + xb.z*yb.z + xb.w*yb.w;
        #pragma unroll
        for (int m = 16; m >= 1; m >>= 1) p += __shfl_xor(p, m, 32);

        const float s = partial[(size_t)row * CSLICE + 0]
                      + partial[(size_t)row * CSLICE + 1]
                      + partial[(size_t)row * CSLICE + 2]
                      + partial[(size_t)row * CSLICE + 3];
        const float pos = p * invn[row] * invn[j] * 10.0f;
        acc += __logf(s) - pos;
    }
    if (lane == 0) red[w] = acc;
    __syncthreads();
    if (tid == 0) {
        float tot = 0.f;
        #pragma unroll
        for (int k = 0; k < 8; ++k) tot += red[k];
        out[0] = tot / (float)NROWS;
    }
}

// ---------------------------------------------------------------------------
extern "C" void kernel_launch(void* const* d_in, const int* in_sizes, int n_in,
                              void* d_out, int out_size, void* d_ws, size_t ws_size,
                              hipStream_t stream) {
    (void)in_sizes; (void)n_in; (void)out_size; (void)ws_size;
    const float* z1 = (const float*)d_in[0];
    const float* z2 = (const float*)d_in[1];

    // workspace layout:
    //   [0, 4MB)          : zn bf16 (8192 x 256, row-major, as u32 dwords)
    //   [4MB, 4MB+32KB)   : 1/norm per row (8192 f32)
    //   [.., +128KB)      : exp-sum partials (8192 x CSLICE f32)
    uint32_t* znbf    = (uint32_t*)d_ws;
    float*    invn    = (float*)((char*)d_ws + (size_t)NROWS * DWPR * 4);
    float*    partial = invn + NROWS;
    float*    out     = (float*)d_out;

    ntx_normalize<<<dim3(NROWS / 8), dim3(256), 0, stream>>>(z1, z2, znbf, invn);
    ntx_gemm_expsum<<<dim3(64, CSLICE), dim3(256), 0, stream>>>(znbf, partial);
    ntx_finalize<<<dim3(1), dim3(256), 0, stream>>>(z1, z2, invn, partial, out);
}
